// DiffGraphormer_11029476016660
// MI455X (gfx1250) — matrix operand order repaired and checked by reading, not compiled, not run on hardware
//
#include <hip/hip_runtime.h>
#include <hip/hip_bf16.h>

typedef __attribute__((ext_vector_type(16))) _Float16 v16h;
typedef __attribute__((ext_vector_type(8)))  float    v8f;

#define HEADS 8
#define CPH   16
#define HID   128
#define NTILES (HID / 16)   // 8 N-tiles of 16 columns

// ---------------------------------------------------------------------------
// out[M,HID] = (A [- Asub]) @ W + bias via v_wmma_f32_16x16x32_f16.
//
// Block = 256 threads = 8 waves. The whole W ([K,128] f32) is converted to
// f16 and staged in LDS, pre-swizzled into the ISA B-fragment layout
// (32x16 bf/f16 B: lane = (n&15) + 16*((kin>>3)&1), elem e = (kin&7) + 8*(kin>=16)) so each
// lane reads its 16-half fragment as one contiguous 32B LDS access.
// Each wave owns a 16-row M-slab and produces all 8 N-tiles (16x128 output),
// reusing its A fragment 8x per K-step. K is a template param so the K-loop
// fully unrolls: K=128 -> 32 straight-line WMMAs, K=64 -> 16.
// ---------------------------------------------------------------------------
template <int K, bool DIFF>
__global__ __launch_bounds__(256)
void gemm_wmma_f16(const float* __restrict__ A, const float* __restrict__ Asub,
                   const float* __restrict__ W, const float* __restrict__ bias,
                   float* __restrict__ out, int M)
{
    __shared__ _Float16 wlds[K * HID];

    // ---- cooperative W -> LDS (f16, B-fragment swizzle) ----
    for (int idx = threadIdx.x; idx < K * HID; idx += 256) {
        const int k   = idx >> 7;          // idx / 128
        const int n   = idx & (HID - 1);   // idx % 128
        const int k0b = k >> 5;            // 32-wide K block
        const int kin = k & 31;
        const int lhi = (kin >> 3) & 1;
        const int e   = (kin & 7) | ((kin >> 4) << 3);
        const int nt  = n >> 4;
        const int ln  = (n & 15) + 16 * lhi;
        wlds[(((k0b * NTILES + nt) * 32) + ln) * 16 + e] = (_Float16)W[idx];
    }
    __syncthreads();

    const int lane  = threadIdx.x & 31;
    const int wave  = threadIdx.x >> 5;
    const int tileM = blockIdx.x * 8 + wave;
    if (tileM >= (M >> 4)) return;         // uniform per wave (after sync)

    const int  lhi  = lane >> 4;           // K-half selector for A
    const int  msub = lane & 15;
    const long aRow = (long)tileM * 16 + msub;
    const float* __restrict__ aPtr = A + aRow * (long)K;
    const float* __restrict__ sPtr = DIFF ? (Asub + aRow * (long)K) : nullptr;

    v8f acc[NTILES];
#pragma unroll
    for (int nt = 0; nt < NTILES; ++nt)
        acc[nt] = (v8f){0.f, 0.f, 0.f, 0.f, 0.f, 0.f, 0.f, 0.f};

#pragma unroll
    for (int k0b = 0; k0b < K / 32; ++k0b) {
        const int k0  = k0b * 32;
        // A fragment: lane holds row msub; elems 0..7 -> K = k0+8*lhi+e,
        // elems 8..15 -> K = k0+16+8*lhi+e (ISA 16-bit A 16x32 layout).
        const int ka0 = k0 + 8 * lhi;
        const int ka1 = k0 + 16 + 8 * lhi;
        v16h a;
#pragma unroll
        for (int e = 0; e < 8; ++e) {
            float f0 = aPtr[ka0 + e];
            float f1 = aPtr[ka1 + e];
            if (DIFF) { f0 -= sPtr[ka0 + e]; f1 -= sPtr[ka1 + e]; }
            a[e]     = (_Float16)f0;
            a[e + 8] = (_Float16)f1;
        }
#pragma unroll
        for (int nt = 0; nt < NTILES; ++nt) {
            const v16h b = *reinterpret_cast<const v16h*>(
                &wlds[(((k0b * NTILES + nt) * 32) + lane) * 16]);
            acc[nt] = __builtin_amdgcn_wmma_f32_16x16x32_f16(
                          false, a, false, b, (short)0, acc[nt], false, false);
        }
    }

#pragma unroll
    for (int nt = 0; nt < NTILES; ++nt) {
        const int colg = nt * 16 + msub;           // D column for this lane
        const float bv = bias[colg];
#pragma unroll
        for (int r = 0; r < 8; ++r) {
            const long row = (long)tileM * 16 + r + 8 * lhi;  // D row layout
            out[row * (long)HID + colg] = acc[nt][r] + bv;
        }
    }
}

// ---------------------------------------------------------------------------
__global__ void zero_f32(float* __restrict__ p, long n)
{
    long i = (long)blockIdx.x * blockDim.x + threadIdx.x;
    if (i < n) p[i] = 0.0f;
}

__device__ __forceinline__ unsigned f32_to_ordkey(float f)
{
    unsigned u = __float_as_uint(f);
    return (u & 0x80000000u) ? ~u : (u | 0x80000000u);
}
__device__ __forceinline__ float ordkey_to_f32(unsigned k)
{
    return (k & 0x80000000u) ? __uint_as_float(k & 0x7fffffffu)
                             : __uint_as_float(~k);
}

// alpha[e,h] = dot(q[dst,h,:], k[src,h,:]) / sqrt(C);  segment-max via u32 key
__global__ void attn_logits(const float* __restrict__ q, const float* __restrict__ k,
                            const int* __restrict__ src, const int* __restrict__ dst,
                            float* __restrict__ alpha, unsigned* __restrict__ amaxKey,
                            long EH)
{
    long t = (long)blockIdx.x * blockDim.x + threadIdx.x;
    if (t >= EH) return;
    const int e = (int)(t >> 3), h = (int)(t & 7);
    const int s = src[e], d = dst[e];
    const float4* qp = reinterpret_cast<const float4*>(q + (long)d * HID + h * CPH);
    const float4* kp = reinterpret_cast<const float4*>(k + (long)s * HID + h * CPH);
    float acc = 0.f;
#pragma unroll
    for (int j = 0; j < 4; ++j) {
        float4 a = qp[j], b = kp[j];
        acc += a.x * b.x + a.y * b.y + a.z * b.z + a.w * b.w;
    }
    acc *= 0.25f;                         // 1/sqrt(16)
    alpha[t] = acc;
    atomicMax(&amaxKey[(long)d * HEADS + h], f32_to_ordkey(acc));
}

// a_exp = exp(alpha - max);  segment-sum via float atomics (in-place on alpha)
__global__ void attn_exp(const int* __restrict__ dst, const unsigned* __restrict__ amaxKey,
                         float* __restrict__ alpha, float* __restrict__ asum, long EH)
{
    long t = (long)blockIdx.x * blockDim.x + threadIdx.x;
    if (t >= EH) return;
    const int e = (int)(t >> 3), h = (int)(t & 7);
    const int d = dst[e];
    const float amax = ordkey_to_f32(amaxKey[(long)d * HEADS + h]);
    const float ex = __expf(alpha[t] - amax);
    alpha[t] = ex;
    atomicAdd(&asum[(long)d * HEADS + h], ex);
}

// msg = (v[src] + enc[e]) * attn; scatter-add into agg[dst]
__global__ void attn_msg(const int* __restrict__ src, const int* __restrict__ dst,
                         const float* __restrict__ aexp, const float* __restrict__ asum,
                         const float* __restrict__ v, const float* __restrict__ enc,
                         float* __restrict__ agg, long EH)
{
    long t = (long)blockIdx.x * blockDim.x + threadIdx.x;
    if (t >= EH) return;
    const int e = (int)(t >> 3), h = (int)(t & 7);
    const int s = src[e], d = dst[e];
    const float attn = aexp[t] / (asum[(long)d * HEADS + h] + 1e-16f);
    const float* vp = v   + (long)s * HID + h * CPH;
    const float* ep = enc + (long)e * HID + h * CPH;
    float* ap = agg + (long)d * HID + h * CPH;
#pragma unroll
    for (int i = 0; i < CPH; ++i)
        atomicAdd(&ap[i], (vp[i] + ep[i]) * attn);
}

__global__ void add_inplace(float* __restrict__ dstp, const float* __restrict__ srcp, long n)
{
    long i = (long)blockIdx.x * blockDim.x + threadIdx.x;
    if (i < n) dstp[i] += srcp[i];
}

// out[e,:] = (x_trans[src] + x_trans[dst]) @ Wc + bc, Wc cached in LDS
__global__ __launch_bounds__(256)
void edge_classifier(const int* __restrict__ src, const int* __restrict__ dst,
                     const float* __restrict__ xt, const float* __restrict__ Wc,
                     const float* __restrict__ bc, float* __restrict__ out, int E)
{
    __shared__ float wcs[HID * 2];
    __shared__ float bcs[2];
    for (int i = threadIdx.x; i < HID * 2; i += blockDim.x) wcs[i] = Wc[i];
    if (threadIdx.x < 2) bcs[threadIdx.x] = bc[threadIdx.x];
    __syncthreads();

    const int e = blockIdx.x * blockDim.x + threadIdx.x;
    if (e >= E) return;
    const float* r0 = xt + (long)src[e] * HID;
    const float* r1 = xt + (long)dst[e] * HID;
    float o0 = bcs[0], o1 = bcs[1];
#pragma unroll 8
    for (int i = 0; i < HID; ++i) {
        const float val = r0[i] + r1[i];
        o0 += val * wcs[2 * i];
        o1 += val * wcs[2 * i + 1];
    }
    out[(long)e * 2]     = o0;
    out[(long)e * 2 + 1] = o1;
}

// ---------------------------------------------------------------------------
extern "C" void kernel_launch(void* const* d_in, const int* in_sizes, int n_in,
                              void* d_out, int out_size, void* d_ws, size_t ws_size,
                              hipStream_t stream)
{
    const float* x_t   = (const float*)d_in[0];
    // d_in[1] (x_t_dt) is unused by the reference
    const float* ea_t  = (const float*)d_in[2];
    const float* ea_dt = (const float*)d_in[3];
    const int*   eidx  = (const int*)  d_in[4];
    const float* Wn = (const float*)d_in[5],  *bn = (const float*)d_in[6];
    const float* We = (const float*)d_in[7],  *be = (const float*)d_in[8];
    const float* Wq = (const float*)d_in[9],  *bq = (const float*)d_in[10];
    const float* Wk = (const float*)d_in[11], *bk = (const float*)d_in[12];
    const float* Wv = (const float*)d_in[13], *bv = (const float*)d_in[14];
    const float* Ws = (const float*)d_in[15], *bs = (const float*)d_in[16];
    const float* Wc = (const float*)d_in[17], *bc = (const float*)d_in[18];
    float* out = (float*)d_out;

    const int N = in_sizes[0] / HID;          // 50000
    const int E = in_sizes[4] / 2;            // 1000000
    const int* src = eidx;                    // row 0 of edge_index
    const int* dst = eidx + E;                // row 1

    // ---- workspace layout (all 4-byte units) ----
    float* base   = (float*)d_ws;
    const long NX = (long)N * HID;
    float*    x      = base;
    float*    q      = base + 1 * NX;
    float*    kbuf   = base + 2 * NX;
    float*    vbuf   = base + 3 * NX;
    float*    xtrans = base + 4 * NX;
    float*    agg    = base + 5 * NX;                    // zeroed
    unsigned* amax   = (unsigned*)(base + 6 * NX);       // zeroed (N*HEADS)
    float*    asum   = base + 6 * NX + (long)N * HEADS;  // zeroed (N*HEADS)
    float*    alpha  = asum + (long)N * HEADS;           // E*HEADS (reused as a_exp)
    float*    enc    = alpha + (long)E * HEADS;          // E*HID
    (void)ws_size; (void)n_in; (void)out_size;

    const int GT = 256;
    const int nodeTiles = N / 16;             // 3125
    const int edgeTiles = E / 16;             // 62500
    const unsigned nodeBlks = (unsigned)((nodeTiles + 7) / 8);
    const unsigned edgeBlks = (unsigned)((edgeTiles + 7) / 8);

    // 1) node encoder: x = x_t @ Wn + bn
    gemm_wmma_f16<128, false><<<nodeBlks, 256, 0, stream>>>(x_t, nullptr, Wn, bn, x, N);
    // 2) edge diff encoder: enc = (ea_dt - ea_t) @ We + be
    gemm_wmma_f16<64, true><<<edgeBlks, 256, 0, stream>>>(ea_dt, ea_t, We, be, enc, E);
    // 3) q / k / v / skip projections from x
    gemm_wmma_f16<128, false><<<nodeBlks, 256, 0, stream>>>(x, nullptr, Wq, bq, q,      N);
    gemm_wmma_f16<128, false><<<nodeBlks, 256, 0, stream>>>(x, nullptr, Wk, bk, kbuf,   N);
    gemm_wmma_f16<128, false><<<nodeBlks, 256, 0, stream>>>(x, nullptr, Wv, bv, vbuf,   N);
    gemm_wmma_f16<128, false><<<nodeBlks, 256, 0, stream>>>(x, nullptr, Ws, bs, xtrans, N);
    // 4) zero agg + amax keys + asum (contiguous region)
    {
        long n = NX + 2L * N * HEADS;
        zero_f32<<<(unsigned)((n + GT - 1) / GT), GT, 0, stream>>>(agg, n);
    }
    const long EH = (long)E * HEADS;
    const unsigned ehBlocks = (unsigned)((EH + GT - 1) / GT);
    // 5) attention logits + segment max
    attn_logits<<<ehBlocks, GT, 0, stream>>>(q, kbuf, src, dst, alpha, amax, EH);
    // 6) exp + segment sum
    attn_exp<<<ehBlocks, GT, 0, stream>>>(dst, amax, alpha, asum, EH);
    // 7) weighted message scatter-add
    attn_msg<<<ehBlocks, GT, 0, stream>>>(src, dst, alpha, asum, vbuf, enc, agg, EH);
    // 8) x_trans = skip + agg
    add_inplace<<<(unsigned)((NX + GT - 1) / GT), GT, 0, stream>>>(xtrans, agg, NX);
    // 9) per-edge classifier
    edge_classifier<<<(unsigned)((E + 255) / 256), 256, 0, stream>>>(
        src, dst, xtrans, Wc, bc, out, E);
}
